// TreeDecoderBatch_73091753443795
// MI455X (gfx1250) — matrix operand order repaired and checked by reading, not compiled, not run on hardware
//
#include <hip/hip_runtime.h>
#include <hip/hip_bf16.h>

typedef __attribute__((ext_vector_type(16))) __bf16 v16bf;
typedef __attribute__((ext_vector_type(8)))  float  v8f;

#define KEXP   16
#define BDIM   4096
#define HDIM   256
#define INDIM  512
#define NROWS  (KEXP * BDIM)          /* 65536 */
#define HOFF   0
#define COFF   ((size_t)NROWS * HDIM) /* 16777216 */
#define LOSSOFF ((size_t)2 * NROWS * HDIM)

// f32 -> bf16 (round-to-nearest-even), packed pair
__device__ __forceinline__ unsigned int pack_bf16(float lo, float hi) {
  unsigned int a = __float_as_uint(lo);
  unsigned int b = __float_as_uint(hi);
  a = (a + 0x7FFFu + ((a >> 16) & 1u)) >> 16;
  b = (b + 0x7FFFu + ((b >> 16) & 1u)) >> 16;
  return (b << 16) | (a & 0xFFFFu);
}

__device__ __forceinline__ float sigmoidf_(float x) { return 1.0f / (1.0f + __expf(-x)); }

// 16-bit A-matrix 16x32 fragment swizzle (ISA 7.12.2):
// lanes 0-15 hold K 0-7,16-23; lanes 16-31 hold K 8-15,24-31 for row rl.
__device__ __forceinline__ void a_swz(int rl, int k, int& lane, int& e) {
  int grp = k >> 3;                 // 0..3
  lane = rl + ((grp & 1) << 4);
  e = k - 8 * ((grp + 1) >> 1);     // halfword index 0..15
}
// 16-bit B-matrix 32x16 fragment swizzle: lane n holds column n (K 0-7,16-23),
// lane n+16 holds K 8-15,24-31, two K per VGPR sequential.
__device__ __forceinline__ void b_swz(int n, int k, int& lane, int& e) {
  int grp = k >> 3;
  lane = n + ((grp & 1) << 4);
  e = k - 8 * ((grp + 1) >> 1);
}

// ---------------------------------------------------------------------------
// Kernel 1: per-expert gate GEMM (bf16 WMMA, f32 accum) + LSTM elementwise
// grid: (4096/64, 256/64, 16); block: 256 (8 waves)
// Each workgroup: 64 rows x 64 H-cols, all 4 gates.
// Wave w: mw=w>>2 (32-row half), ncp=w&3 (16-col subtile): holds i/f/g/o accums.
// ---------------------------------------------------------------------------
__global__ __launch_bounds__(256)
void lstm_gates_kernel(const float* __restrict__ x, const float* __restrict__ h,
                       const float* __restrict__ c,
                       const float* __restrict__ Wih, const float* __restrict__ Whh,
                       const float* __restrict__ bih, const float* __restrict__ bhh,
                       const int* __restrict__ child_index,
                       float* __restrict__ out) {
  __shared__ __align__(32) unsigned char smem[4 * 1024 + 16 * 1024];
  unsigned int* smw = (unsigned int*)smem;
  const v16bf* frag = (const v16bf*)smem;   // 32-byte fragment granules

  const int tid  = threadIdx.x;
  const int lane = tid & 31;
  const int w    = tid >> 5;
  const int mw   = w >> 2;        // 0..1 -> rows [mw*32, mw*32+32)
  const int ncp  = w & 3;         // 0..3 -> H cols [ncp*16, +16)

  const int m0 = blockIdx.x * 64;
  const int n0 = blockIdx.y * 64;
  const int z  = blockIdx.z;              // output slot
  const int ke = child_index[z];          // expert id (identity permutation)

  v8f acc[4][2];
  for (int g = 0; g < 4; ++g)
    for (int i = 0; i < 2; ++i)
      for (int q = 0; q < 8; ++q) acc[g][i][q] = 0.0f;

  for (int phase = 0; phase < 2; ++phase) {
    const float* asrc = (phase == 0) ? (x + ((size_t)ke * BDIM + m0) * INDIM)
                                     : (h + ((size_t)ke * BDIM + m0) * HDIM);
    const float* wsrc = (phase == 0) ? (Wih + (size_t)ke * 4 * HDIM * INDIM)
                                     : (Whh + (size_t)ke * 4 * HDIM * HDIM);
    const int ldk = (phase == 0) ? INDIM : HDIM;

    for (int kk = 0; kk < ldk; kk += 32) {
      __syncthreads();
      // stage A tile: 64 rows x 32 K, fragment order (4 subtiles x 1KB)
      for (int j = 0; j < 4; ++j) {
        int pi = tid + 256 * j;
        int r = pi >> 4, k = (pi & 15) * 2;
        float2 v = *(const float2*)(asrc + (size_t)r * ldk + kk + k);
        int ln, e; a_swz(r & 15, k, ln, e);
        smw[(((r >> 4) * 1024) + ln * 32 + e * 2) >> 2] = pack_bf16(v.x, v.y);
      }
      // stage B tile: 4 gates x 64 cols x 32 K (16 blocks x 1KB at +4KB)
      for (int j = 0; j < 16; ++j) {
        int pi = tid + 256 * j;
        int wr = pi >> 4, k = (pi & 15) * 2;
        int g = wr >> 6, within = wr & 63;
        int wrow = g * HDIM + n0 + within;
        float2 v = *(const float2*)(wsrc + (size_t)wrow * ldk + kk + k);
        int ln, e; b_swz(within & 15, k, ln, e);
        smw[(4096 + (g * 4 + (within >> 4)) * 1024 + ln * 32 + e * 2) >> 2] =
            pack_bf16(v.x, v.y);
      }
      __syncthreads();
      if (kk + 32 < ldk)   // hint next K slice into cache (global_prefetch_b8)
        __builtin_prefetch(asrc + (size_t)(tid >> 2) * ldk + kk + 32, 0, 1);

      v16bf a0 = frag[(mw * 2 + 0) * 32 + lane];
      v16bf a1 = frag[(mw * 2 + 1) * 32 + lane];
      for (int g = 0; g < 4; ++g) {
        v16bf bg = frag[128 + (g * 4 + ncp) * 32 + lane];
        acc[g][0] = __builtin_amdgcn_wmma_f32_16x16x32_bf16(
            false, a0, false, bg, (short)0, acc[g][0], false, false);
        acc[g][1] = __builtin_amdgcn_wmma_f32_16x16x32_bf16(
            false, a1, false, bg, (short)0, acc[g][1], false, false);
      }
    }
  }

  // epilogue: C-tile VGPR j holds M=j (lanes 0-15) / M=j+8 (lanes 16-31), N=lane%16
  const int col = n0 + ncp * 16 + (lane & 15);
  float bsum[4];
  for (int g = 0; g < 4; ++g)
    bsum[g] = bih[ke * 4 * HDIM + g * HDIM + col] + bhh[ke * 4 * HDIM + g * HDIM + col];

  for (int i = 0; i < 2; ++i) {
    for (int j = 0; j < 8; ++j) {
      int rl   = j + ((lane >> 4) << 3);
      int rloc = m0 + mw * 32 + i * 16 + rl;
      size_t rf = (size_t)z * BDIM + rloc;
      float cp = c[((size_t)ke * BDIM + rloc) * HDIM + col];
      float ig = sigmoidf_(acc[0][i][j] + bsum[0]);
      float fg = sigmoidf_(acc[1][i][j] + bsum[1]);
      float gg = tanhf(acc[2][i][j] + bsum[2]);
      float og = sigmoidf_(acc[3][i][j] + bsum[3]);
      float cn = fg * cp + ig * gg;
      float hn = og * tanhf(cn);
      out[HOFF + rf * HDIM + col] = hn;
      out[COFF + rf * HDIM + col] = cn;
    }
  }
}

// ---------------------------------------------------------------------------
// Kernel 2: logits = h @ W_out^T + b_out, streamed online log-sum-exp -> loss
// grid: 65536/64 = 1024; block 256 (8 waves). Wave: mw=w&3 rows, cp=w>>2 cols.
// ---------------------------------------------------------------------------
__global__ __launch_bounds__(256)
void head_ce_kernel(const float* __restrict__ Wout, const float* __restrict__ bout,
                    const int* __restrict__ tval, float* __restrict__ out) {
  __shared__ __align__(32) unsigned char smem[32 * 1024 + 4 * 1024 + 16 * 1024 + 3 * 1024];
  unsigned int* smw = (unsigned int*)smem;
  float* smLog = (float*)(smem + 36 * 1024);          // 64x64 logits tile
  float* smRed = (float*)(smem + 52 * 1024);          // 3 x 256 reduction
  const v16bf* fragA = (const v16bf*)smem;            // 32 blocks (8 ksteps x 4 subtiles)
  const v16bf* fragB = (const v16bf*)(smem + 32 * 1024);
  const float* hsrc = out;                            // h_flat from kernel 1

  const int tid = threadIdx.x, lane = tid & 31, w = tid >> 5;
  const int mw = w & 3, cp = w >> 2;
  const size_t r0 = (size_t)blockIdx.x * 64;

  // stage all A fragments once: 64 rows x 256 K (f32 -> bf16)
  for (int j = 0; j < 32; ++j) {
    int pi = tid + 256 * j;
    int r = pi >> 7, kabs = (pi & 127) * 2;
    float2 v = *(const float2*)(hsrc + (r0 + r) * HDIM + kabs);
    int ks = kabs >> 5, kin = kabs & 31;
    int ln, e; a_swz(r & 15, kin, ln, e);
    smw[(((ks * 4 + (r >> 4)) * 1024) + ln * 32 + e * 2) >> 2] = pack_bf16(v.x, v.y);
  }

  float m = -1e30f, s = 0.0f, tl = 0.0f;
  const int myrow = tid & 63, myseg = tid >> 6;
  const int tv = tval[r0 + myrow];

  for (int nb = 0; nb < 8; ++nb) {
    v8f accB[2];
    for (int jj = 0; jj < 2; ++jj)
      for (int q = 0; q < 8; ++q) accB[jj][q] = 0.0f;

    for (int ks = 0; ks < 8; ++ks) {
      __syncthreads();
      // stage B slice: 64 classes x 32 K
      for (int j = 0; j < 4; ++j) {
        int pi = tid + 256 * j;
        int wr = pi >> 4, k = (pi & 15) * 2;
        float2 v = *(const float2*)(Wout + (size_t)(nb * 64 + wr) * HDIM + ks * 32 + k);
        int ln, e; b_swz(wr & 15, k, ln, e);
        smw[(32 * 1024 + (wr >> 4) * 1024 + ln * 32 + e * 2) >> 2] = pack_bf16(v.x, v.y);
      }
      __syncthreads();
      v16bf af = fragA[(ks * 4 + mw) * 32 + lane];
      for (int jj = 0; jj < 2; ++jj) {
        v16bf bf = fragB[(cp * 2 + jj) * 32 + lane];
        accB[jj] = __builtin_amdgcn_wmma_f32_16x16x32_bf16(
            false, af, false, bf, (short)0, accB[jj], false, false);
      }
    }
    // logits tile (+bias) -> LDS
    for (int jj = 0; jj < 2; ++jj) {
      int ct = (cp * 2 + jj) * 16 + (lane & 15);
      float bb = bout[nb * 64 + ct];
      for (int q = 0; q < 8; ++q) {
        int rl = mw * 16 + q + ((lane >> 4) << 3);
        smLog[rl * 64 + ct] = accB[jj][q] + bb;
      }
    }
    __syncthreads();
    // online log-sum-exp over this 64-class block (4 segments/row)
    for (int q = 0; q < 16; ++q) {
      int cin = myseg * 16 + q;
      float v = smLog[myrow * 64 + cin];
      float nm = fmaxf(m, v);
      s = s * __expf(m - nm) + __expf(v - nm);
      m = nm;
      if (nb * 64 + cin == tv) tl = v;
    }
  }

  // merge segment partials per row
  smRed[tid] = m; smRed[256 + tid] = s; smRed[512 + tid] = tl;
  __syncthreads();
  if (tid < 64) {
    float M = -1e30f;
    for (int sg = 0; sg < 4; ++sg) M = fmaxf(M, smRed[sg * 64 + tid]);
    float S = 0.0f, T = 0.0f;
    for (int sg = 0; sg < 4; ++sg) {
      S += smRed[256 + sg * 64 + tid] * __expf(smRed[sg * 64 + tid] - M);
      T += smRed[512 + sg * 64 + tid];
    }
    out[LOSSOFF + r0 + tid] = __logf(S) + M - T;
  }
}

extern "C" void kernel_launch(void* const* d_in, const int* in_sizes, int n_in,
                              void* d_out, int out_size, void* d_ws, size_t ws_size,
                              hipStream_t stream) {
  const float* x    = (const float*)d_in[0];
  const float* h    = (const float*)d_in[1];
  const float* c    = (const float*)d_in[2];
  const float* Wih  = (const float*)d_in[3];
  const float* Whh  = (const float*)d_in[4];
  const float* bih  = (const float*)d_in[5];
  const float* bhh  = (const float*)d_in[6];
  const float* Wout = (const float*)d_in[7];
  const float* bout = (const float*)d_in[8];
  const int*   cidx = (const int*)d_in[9];
  const int*   tval = (const int*)d_in[10];
  float* out = (float*)d_out;

  dim3 gA(BDIM / 64, HDIM / 64, KEXP);   // 64 x 4 x 16 workgroups
  hipLaunchKernelGGL(lstm_gates_kernel, gA, dim3(256), 0, stream,
                     x, h, c, Wih, Whh, bih, bhh, cidx, out);
  dim3 gB(NROWS / 64);                   // 1024 workgroups
  hipLaunchKernelGGL(head_ce_kernel, gB, dim3(256), 0, stream,
                     Wout, bout, tval, out);
}
